// FusedMoEModularKernel_10350871183626
// MI455X (gfx1250) — matrix operand order repaired and checked by reading, not compiled, not run on hardware
//
#include <hip/hip_runtime.h>

#define MDIM 4096
#define KDIM 2048
#define NDIM 5632
#define EEXP 8
#define TOPKK 2
#define CAP  (MDIM * TOPKK)          // 8192 token-expert pairs
#define LDAB 40                      // LDS row stride in bf16 (80B = 20 banks, conflict-free)

typedef __attribute__((ext_vector_type(16))) __bf16 v16bf;
typedef __attribute__((ext_vector_type(2)))  __bf16 v2bf;
typedef __attribute__((ext_vector_type(8)))  float  v8f;

union BFrag { v16bf bf; uint4 q[2]; };

#if __has_builtin(__builtin_amdgcn_cvt_pk_bf16_f32)
__device__ __forceinline__ unsigned int pk2(float lo, float hi) {
  union { v2bf b; unsigned int u; } c;
  c.b = __builtin_amdgcn_cvt_pk_bf16_f32(lo, hi);   // v_cvt_pk_bf16_f32
  return c.u;
}
#else
// round-half-up to bf16, packed pair via one v_perm_b32:
// result = { (hi+0x8000)[31:16], (lo+0x8000)[31:16] }
__device__ __forceinline__ unsigned int pk2(float lo, float hi) {
  unsigned int ua = __float_as_uint(lo) + 0x8000u;
  unsigned int ub = __float_as_uint(hi) + 0x8000u;
  // sel 0x07060302: byte0=src1.b2, byte1=src1.b3, byte2=src0.b2, byte3=src0.b3
  return __builtin_amdgcn_perm(ub, ua, 0x07060302u);
}
#endif

// ---------------------------------------------------------------- routing ---
__global__ __launch_bounds__(256) void moe_route(const int* __restrict__ ids,
                                                 int* __restrict__ counts,
                                                 int* __restrict__ list) {
  int t = blockIdx.x * 256 + threadIdx.x;           // pair id = m*TOPK + s
  if (t < CAP) {
    int e = ids[t];
    int pos = atomicAdd(&counts[e], 1);
    list[e * CAP + pos] = t;
  }
}

// ------------------------------------------------------- GEMM1 + SiLU*up ---
// C tile: 128 pair-rows x (64 gate cols + 64 up cols). 8 waves = 4 row x 2 col
// groups; each wave: 2x4 wmma accumulators (g at j=0..1, u at j=2..3).
__global__ __launch_bounds__(256) void moe_gemm1(const float* __restrict__ hs,
                                                 const float* __restrict__ w1,
                                                 const int* __restrict__ counts,
                                                 const int* __restrict__ list,
                                                 unsigned short* __restrict__ aBuf) {
  const int e   = blockIdx.z;
  const int cnt = counts[e];
  const int tm  = blockIdx.x * 128;
  if (tm >= cnt) return;                            // dynamic early-exit
  const int n0  = blockIdx.y * 64;

  __shared__ __align__(16) unsigned short ldsA[128 * LDAB];
  __shared__ __align__(16) unsigned short ldsB[128 * LDAB];
  __shared__ int ldsIdx[128];

  const int t  = threadIdx.x;
  const int lr = t >> 1;                            // loader row 0..127
  const int hc = (t & 1) * 16;                      // loader K sub-chunk

  if (t < 128) {
    int gi = tm + t; if (gi > cnt - 1) gi = cnt - 1;
    ldsIdx[t] = list[e * CAP + gi];
  }

  int giA = tm + lr; if (giA > cnt - 1) giA = cnt - 1;
  const int pairA = list[e * CAP + giA];
  const float* srcA = hs + (size_t)(pairA >> 1) * KDIM + hc;

  const int w1row = (lr < 64) ? (n0 + lr) : (NDIM + n0 + (lr - 64));
  const float* srcB = w1 + (size_t)e * (2ull * NDIM * KDIM)
                         + (size_t)w1row * KDIM + hc;

  float4 aR[4], bR[4];
#pragma unroll
  for (int q = 0; q < 4; ++q) {
    aR[q] = *(const float4*)(srcA + 4 * q);
    bR[q] = *(const float4*)(srcB + 4 * q);
  }

  const int wave = t >> 5, wr = wave >> 1, wc = wave & 1;
  const int lane = t & 31, half = lane >> 4, lm = lane & 15;

  v8f acc[2][4];
#pragma unroll
  for (int i = 0; i < 2; ++i)
#pragma unroll
    for (int j = 0; j < 4; ++j)
#pragma unroll
      for (int q = 0; q < 8; ++q) acc[i][j][q] = 0.0f;

  for (int kt = 0; kt < KDIM; kt += 32) {
    __syncthreads();
    {
      uint4* dA = (uint4*)&ldsA[lr * LDAB + hc];
      dA[0] = make_uint4(pk2(aR[0].x, aR[0].y), pk2(aR[0].z, aR[0].w),
                         pk2(aR[1].x, aR[1].y), pk2(aR[1].z, aR[1].w));
      dA[1] = make_uint4(pk2(aR[2].x, aR[2].y), pk2(aR[2].z, aR[2].w),
                         pk2(aR[3].x, aR[3].y), pk2(aR[3].z, aR[3].w));
      uint4* dB = (uint4*)&ldsB[lr * LDAB + hc];
      dB[0] = make_uint4(pk2(bR[0].x, bR[0].y), pk2(bR[0].z, bR[0].w),
                         pk2(bR[1].x, bR[1].y), pk2(bR[1].z, bR[1].w));
      dB[1] = make_uint4(pk2(bR[2].x, bR[2].y), pk2(bR[2].z, bR[2].w),
                         pk2(bR[3].x, bR[3].y), pk2(bR[3].z, bR[3].w));
    }
    __syncthreads();
    if (kt + 32 < KDIM) {                           // reg prefetch next K-step
#pragma unroll
      for (int q = 0; q < 4; ++q) {
        aR[q] = *(const float4*)(srcA + kt + 32 + 4 * q);
        bR[q] = *(const float4*)(srcB + kt + 32 + 4 * q);
      }
      __builtin_prefetch(srcB + kt + 32 + 256, 0, 0);   // global_prefetch_b8,
      __builtin_prefetch(srcA + kt + 32 + 256, 0, 0);   // ~8 K-steps ahead
    }
    BFrag fA[2], fB[4];
#pragma unroll
    for (int i = 0; i < 2; ++i) {
      int base = (32 * wr + 16 * i + lm) * LDAB + half * 8;
      fA[i].q[0] = *(const uint4*)&ldsA[base];
      fA[i].q[1] = *(const uint4*)&ldsA[base + 16];
    }
#pragma unroll
    for (int j = 0; j < 4; ++j) {
      int c0 = (j < 2) ? (32 * wc + 16 * j) : (64 + 32 * wc + 16 * (j - 2));
      int rb = (c0 + lm) * LDAB + half * 8;
      fB[j].q[0] = *(const uint4*)&ldsB[rb];
      fB[j].q[1] = *(const uint4*)&ldsB[rb + 16];
    }
#pragma unroll
    for (int i = 0; i < 2; ++i)
#pragma unroll
      for (int j = 0; j < 4; ++j)
        acc[i][j] = __builtin_amdgcn_wmma_f32_16x16x32_bf16(
            false, fA[i].bf, false, fB[j].bf, (short)0, acc[i][j], false, false);
  }

  // fused SiLU(g) * u epilogue, scatter bf16 rows by pair id
  const int rl0 = 32 * wr + half * 8;
  if (tm + 128 <= cnt) {                            // full tile: no guards
#pragma unroll
    for (int i = 0; i < 2; ++i)
#pragma unroll
      for (int jj = 0; jj < 2; ++jj) {
        const int col = n0 + 32 * wc + 16 * jj + lm;
#pragma unroll
        for (int v = 0; v < 8; ++v) {
          const int pair = ldsIdx[rl0 + 16 * i + v];
          const float g = acc[i][jj][v];
          const float u = acc[i][jj + 2][v];
          const float s = g / (1.0f + __expf(-g)) * u;
          aBuf[(size_t)pair * NDIM + col] = (unsigned short)pk2(s, s);
        }
      }
  } else {
#pragma unroll
    for (int i = 0; i < 2; ++i)
#pragma unroll
      for (int jj = 0; jj < 2; ++jj) {
        const int col = n0 + 32 * wc + 16 * jj + lm;
#pragma unroll
        for (int v = 0; v < 8; ++v) {
          const int rl = rl0 + 16 * i + v;
          if (tm + rl < cnt) {
            const int pair = ldsIdx[rl];
            const float g = acc[i][jj][v];
            const float u = acc[i][jj + 2][v];
            const float s = g / (1.0f + __expf(-g)) * u;
            aBuf[(size_t)pair * NDIM + col] = (unsigned short)pk2(s, s);
          }
        }
      }
  }
}

// ------------------------------------------------------------------ GEMM2 ---
// t2[pair, 0:K] = a[pair, 0:N] @ w2[e].T ; 128x128 tile, waves 4x2 (32x64).
__global__ __launch_bounds__(256) void moe_gemm2(const unsigned short* __restrict__ aBuf,
                                                 const float* __restrict__ w2,
                                                 const int* __restrict__ counts,
                                                 const int* __restrict__ list,
                                                 float* __restrict__ t2) {
  const int e   = blockIdx.z;
  const int cnt = counts[e];
  const int tm  = blockIdx.x * 128;
  if (tm >= cnt) return;
  const int k0  = blockIdx.y * 128;

  __shared__ __align__(16) unsigned short ldsA[128 * LDAB];
  __shared__ __align__(16) unsigned short ldsB[128 * LDAB];
  __shared__ int ldsIdx[128];

  const int t  = threadIdx.x;
  const int lr = t >> 1;
  const int hc = (t & 1) * 16;

  if (t < 128) {
    int gi = tm + t; if (gi > cnt - 1) gi = cnt - 1;
    ldsIdx[t] = list[e * CAP + gi];
  }

  int giA = tm + lr; if (giA > cnt - 1) giA = cnt - 1;
  const int pairA = list[e * CAP + giA];
  const unsigned short* srcA = aBuf + (size_t)pairA * NDIM + hc;   // bf16 rows
  const float* srcB = w2 + (size_t)e * ((size_t)KDIM * NDIM)
                         + (size_t)(k0 + lr) * NDIM + hc;

  uint4  aR[2];
  float4 bR[4];
  aR[0] = *(const uint4*)(srcA);
  aR[1] = *(const uint4*)(srcA + 8);
#pragma unroll
  for (int q = 0; q < 4; ++q) bR[q] = *(const float4*)(srcB + 4 * q);

  const int wave = t >> 5, wr = wave >> 1, wc = wave & 1;
  const int lane = t & 31, half = lane >> 4, lm = lane & 15;

  v8f acc[2][4];
#pragma unroll
  for (int i = 0; i < 2; ++i)
#pragma unroll
    for (int j = 0; j < 4; ++j)
#pragma unroll
      for (int q = 0; q < 8; ++q) acc[i][j][q] = 0.0f;

  for (int nt = 0; nt < NDIM; nt += 32) {
    __syncthreads();
    {
      uint4* dA = (uint4*)&ldsA[lr * LDAB + hc];
      dA[0] = aR[0];
      dA[1] = aR[1];
      uint4* dB = (uint4*)&ldsB[lr * LDAB + hc];
      dB[0] = make_uint4(pk2(bR[0].x, bR[0].y), pk2(bR[0].z, bR[0].w),
                         pk2(bR[1].x, bR[1].y), pk2(bR[1].z, bR[1].w));
      dB[1] = make_uint4(pk2(bR[2].x, bR[2].y), pk2(bR[2].z, bR[2].w),
                         pk2(bR[3].x, bR[3].y), pk2(bR[3].z, bR[3].w));
    }
    __syncthreads();
    if (nt + 32 < NDIM) {
      aR[0] = *(const uint4*)(srcA + nt + 32);
      aR[1] = *(const uint4*)(srcA + nt + 32 + 8);
#pragma unroll
      for (int q = 0; q < 4; ++q)
        bR[q] = *(const float4*)(srcB + nt + 32 + 4 * q);
      __builtin_prefetch(srcB + nt + 32 + 256, 0, 0);
      __builtin_prefetch(srcA + nt + 32 + 256, 0, 0);
    }
    BFrag fA[2], fB[4];
#pragma unroll
    for (int i = 0; i < 2; ++i) {
      int base = (32 * wr + 16 * i + lm) * LDAB + half * 8;
      fA[i].q[0] = *(const uint4*)&ldsA[base];
      fA[i].q[1] = *(const uint4*)&ldsA[base + 16];
    }
#pragma unroll
    for (int j = 0; j < 4; ++j) {
      int rb = (64 * wc + 16 * j + lm) * LDAB + half * 8;
      fB[j].q[0] = *(const uint4*)&ldsB[rb];
      fB[j].q[1] = *(const uint4*)&ldsB[rb + 16];
    }
#pragma unroll
    for (int i = 0; i < 2; ++i)
#pragma unroll
      for (int j = 0; j < 4; ++j)
        acc[i][j] = __builtin_amdgcn_wmma_f32_16x16x32_bf16(
            false, fA[i].bf, false, fB[j].bf, (short)0, acc[i][j], false, false);
  }

  const int rl0 = 32 * wr + half * 8;
  if (tm + 128 <= cnt) {                            // full tile: no guards
#pragma unroll
    for (int i = 0; i < 2; ++i)
#pragma unroll
      for (int j = 0; j < 4; ++j) {
        const int col = k0 + 64 * wc + 16 * j + lm;
#pragma unroll
        for (int v = 0; v < 8; ++v) {
          const int pair = ldsIdx[rl0 + 16 * i + v];
          t2[(size_t)pair * KDIM + col] = acc[i][j][v];
        }
      }
  } else {
#pragma unroll
    for (int i = 0; i < 2; ++i)
#pragma unroll
      for (int j = 0; j < 4; ++j) {
        const int col = k0 + 64 * wc + 16 * j + lm;
#pragma unroll
        for (int v = 0; v < 8; ++v) {
          const int rl = rl0 + 16 * i + v;
          if (tm + rl < cnt) {
            const int pair = ldsIdx[rl];
            t2[(size_t)pair * KDIM + col] = acc[i][j][v];
          }
        }
      }
  }
}

// ---------------------------------------------------------------- reduce ----
// out[m,k] = w[m,0]*t2[2m,k] + w[m,1]*t2[2m+1,k]  (no atomics, deterministic)
__global__ __launch_bounds__(256) void moe_reduce(const float* __restrict__ t2,
                                                  const float* __restrict__ tw,
                                                  float* __restrict__ out) {
  const size_t i4 = (size_t)blockIdx.x * 256 + threadIdx.x;   // over M*K/4
  const size_t m  = i4 / (KDIM / 4);
  const size_t k  = (i4 % (KDIM / 4)) * 4;
  const float w0 = tw[2 * m], w1 = tw[2 * m + 1];
  const float4 x = *(const float4*)&t2[(2 * m)     * (size_t)KDIM + k];
  const float4 y = *(const float4*)&t2[(2 * m + 1) * (size_t)KDIM + k];
  float4 r;
  r.x = w0 * x.x + w1 * y.x;
  r.y = w0 * x.y + w1 * y.y;
  r.z = w0 * x.z + w1 * y.z;
  r.w = w0 * x.w + w1 * y.w;
  *(float4*)&out[m * (size_t)KDIM + k] = r;
}

// ---------------------------------------------------------------- launch ----
extern "C" void kernel_launch(void* const* d_in, const int* in_sizes, int n_in,
                              void* d_out, int out_size, void* d_ws, size_t ws_size,
                              hipStream_t stream) {
  const float* hs  = (const float*)d_in[0];
  const float* w1  = (const float*)d_in[1];
  const float* w2  = (const float*)d_in[2];
  const float* tw  = (const float*)d_in[3];
  const int*   ids = (const int*)d_in[4];
  float* out = (float*)d_out;

  char* ws = (char*)d_ws;
  constexpr size_t OFF_CNT  = 0;                                   // E ints
  constexpr size_t OFF_LIST = 256;                                 // E*CAP ints
  constexpr size_t OFF_A    = 512 * 1024;                          // CAP*N bf16
  constexpr size_t OFF_T2   = OFF_A + (size_t)CAP * NDIM * 2;      // CAP*K f32
  int* counts = (int*)(ws + OFF_CNT);
  int* list   = (int*)(ws + OFF_LIST);
  unsigned short* aBuf = (unsigned short*)(ws + OFF_A);
  float* t2 = (float*)(ws + OFF_T2);

  hipMemsetAsync(counts, 0, EEXP * sizeof(int), stream);
  moe_route<<<CAP / 256, 256, 0, stream>>>(ids, counts, list);
  moe_gemm1<<<dim3(CAP / 128, NDIM / 64, EEXP), 256, 0, stream>>>(hs, w1, counts, list, aBuf);
  moe_gemm2<<<dim3(CAP / 128, KDIM / 128, EEXP), 256, 0, stream>>>(aBuf, w2, counts, list, t2);
  moe_reduce<<<(MDIM * (KDIM / 4)) / 256, 256, 0, stream>>>(t2, tw, out);
}
